// Top2Gate_38319698214957
// MI455X (gfx1250) — matrix operand order repaired and checked by reading, not compiled, not run on hardware
//
#include <hip/hip_runtime.h>
#include <hip/hip_bf16.h>
#include <float.h>

typedef __attribute__((ext_vector_type(16))) _Float16 v16h;
typedef __attribute__((ext_vector_type(8)))  float    v8f;
typedef __attribute__((ext_vector_type(4)))  float    v4f;

#define T_TOK 8192
#define DDIM  2048
#define NEXP  64
#define CAPC  256

// ---------------------------------------------------------------------------
// K0: W2h[e][k] = (centroids[e]/||centroids[e]||) . w_red[:,k]  -> f16
//     (the reference's 1.5 renorm then F.normalize(eps=1e-4) cancels to a
//      plain unit-normalize since max(1.5, 1e-4) == 1.5)
// Also zeroes the me_sum accumulator for l_aux.
// ---------------------------------------------------------------------------
__global__ void prep_kernel(const float* __restrict__ w_red,
                            const float* __restrict__ cent,
                            _Float16* __restrict__ w2h,
                            float* __restrict__ me_sum) {
  int i = blockIdx.x * blockDim.x + threadIdx.x;
  if (blockIdx.x == 0 && threadIdx.x < NEXP) me_sum[threadIdx.x] = 0.f;
  if (i >= NEXP * DDIM) return;
  int e = i >> 11;           // /2048
  int k = i & 2047;
  float c0 = cent[e * 4 + 0], c1 = cent[e * 4 + 1];
  float c2 = cent[e * 4 + 2], c3 = cent[e * 4 + 3];
  float inv = 1.f / sqrtf(c0 * c0 + c1 * c1 + c2 * c2 + c3 * c3);
  float v = (c0 * w_red[k] + c1 * w_red[DDIM + k] +
             c2 * w_red[2 * DDIM + k] + c3 * w_red[3 * DDIM + k]) * inv;
  w2h[(size_t)e * DDIM + k] = (_Float16)v;
}

// ---------------------------------------------------------------------------
// K1: fused gating GEMM logits = x @ W2^T via v_wmma_f32_16x16x32_f16.
// One wave = 16 tokens x 64 experts (4 N-tiles). Then per-token softmax,
// top-2 selection, and per-expert gate-mean accumulation for l_aux.
// ---------------------------------------------------------------------------
__global__ __launch_bounds__(128) void gate_kernel(
    const float* __restrict__ x, const _Float16* __restrict__ w2h,
    int* __restrict__ idx1, int* __restrict__ idx2,
    float* __restrict__ g1raw, float* __restrict__ g2raw,
    float* __restrict__ me_sum) {
  __shared__ float lgt[4][16][NEXP + 1];   // +1: bank-conflict-free scans
  __shared__ float sme[NEXP];
  const int lane = threadIdx.x & 31;
  const int wv   = threadIdx.x >> 5;
  if (threadIdx.x < NEXP) sme[threadIdx.x] = 0.f;
  __syncthreads();

  const int tile  = blockIdx.x * 4 + wv;        // 16-token tile id
  const int arow  = tile * 16 + (lane & 15);    // A-matrix row (token)
  const int khalf = (lane >> 4) * 16;           // A K sub-offset (contiguous k = 16h + s)
  const float*    xr    = x   + (size_t)arow * DDIM;
  const _Float16* bbase = w2h + (size_t)(lane & 15) * DDIM + (lane >> 4) * 16;

  v8f acc0 = {}, acc1 = {}, acc2 = {}, acc3 = {};
  for (int kc = 0; kc < DDIM; kc += 32) {
    // A tile: convert 16 f32 of this row to the 16x32 f16 WMMA-A layout
    v4f a0 = *(const v4f*)(xr + kc + khalf);
    v4f a1 = *(const v4f*)(xr + kc + khalf + 4);
    v4f a2 = *(const v4f*)(xr + kc + khalf + 8);
    v4f a3 = *(const v4f*)(xr + kc + khalf + 12);
    v16h A;
    A[0]  = (_Float16)a0.x; A[1]  = (_Float16)a0.y;
    A[2]  = (_Float16)a0.z; A[3]  = (_Float16)a0.w;
    A[4]  = (_Float16)a1.x; A[5]  = (_Float16)a1.y;
    A[6]  = (_Float16)a1.z; A[7]  = (_Float16)a1.w;
    A[8]  = (_Float16)a2.x; A[9]  = (_Float16)a2.y;
    A[10] = (_Float16)a2.z; A[11] = (_Float16)a2.w;
    A[12] = (_Float16)a3.x; A[13] = (_Float16)a3.y;
    A[14] = (_Float16)a3.z; A[15] = (_Float16)a3.w;
    // B tiles: 16 contiguous f16 (32B) per lane, one per expert 16-tile
    v16h B0 = *(const v16h*)(bbase + kc);
    v16h B1 = *(const v16h*)(bbase + (size_t)16 * DDIM + kc);
    v16h B2 = *(const v16h*)(bbase + (size_t)32 * DDIM + kc);
    v16h B3 = *(const v16h*)(bbase + (size_t)48 * DDIM + kc);
    acc0 = __builtin_amdgcn_wmma_f32_16x16x32_f16(false, A, false, B0, (short)0, acc0, false, false);
    acc1 = __builtin_amdgcn_wmma_f32_16x16x32_f16(false, A, false, B1, (short)0, acc1, false, false);
    acc2 = __builtin_amdgcn_wmma_f32_16x16x32_f16(false, A, false, B2, (short)0, acc2, false, false);
    acc3 = __builtin_amdgcn_wmma_f32_16x16x32_f16(false, A, false, B3, (short)0, acc3, false, false);
  }
  // C layout: lane holds rows v + 8*(lane>>4), column lane&15
  const int mbase = (lane >> 4) * 8;
  const int ncol  = lane & 15;
#pragma unroll
  for (int v = 0; v < 8; ++v) {
    lgt[wv][mbase + v][ncol]      = acc0[v];
    lgt[wv][mbase + v][16 + ncol] = acc1[v];
    lgt[wv][mbase + v][32 + ncol] = acc2[v];
    lgt[wv][mbase + v][48 + ncol] = acc3[v];
  }
  __syncthreads();

  if (lane < 16) {                       // one lane per token
    const float* rowl = lgt[wv][lane];
    float mx = rowl[0];
#pragma unroll
    for (int e = 1; e < NEXP; ++e) mx = fmaxf(mx, rowl[e]);
    float s = 0.f, v1 = -FLT_MAX, v2 = -FLT_MAX;
    int i1 = 0, i2 = 0;
#pragma unroll
    for (int e = 0; e < NEXP; ++e) {
      float lv = rowl[e];
      s += __expf(lv - mx);
      if (lv > v1)      { v2 = v1; i2 = i1; v1 = lv; i1 = e; }  // first-max tie rule
      else if (lv > v2) { v2 = lv; i2 = e; }
    }
    float invs = 1.f / s;
    int t = tile * 16 + lane;
    idx1[t]  = i1;
    idx2[t]  = i2;
    g1raw[t] = __expf(v1 - mx) * invs;
    g2raw[t] = __expf(v2 - mx) * invs;
#pragma unroll
    for (int e = 0; e < NEXP; ++e)
      atomicAdd(&sme[e], __expf(rowl[e] - mx) * invs);   // gates column sums
  }
  __syncthreads();
  if (threadIdx.x < NEXP) atomicAdd(&me_sum[threadIdx.x], sme[threadIdx.x]);
}

// ---------------------------------------------------------------------------
// K2: per-expert token-order prefix scan (wave32 ballot). loc1 = rank among
// top1 hits; loc2 = splits1[e] + rank among top2 hits. Writes splits as f32
// straight into the output tail.
// ---------------------------------------------------------------------------
__global__ __launch_bounds__(32) void positions_kernel(
    const int* __restrict__ idx1, const int* __restrict__ idx2,
    int* __restrict__ loc1, int* __restrict__ loc2,
    float* __restrict__ splits_out, int* __restrict__ splits1_ws) {
  const int e = blockIdx.x;
  const int lane = threadIdx.x;
  const unsigned below_mask = (1u << lane) - 1u;
  int run = 0;
  for (int base = 0; base < T_TOK; base += 32) {
    int t = base + lane;
    bool m = (idx1[t] == e);
    unsigned bal = __builtin_amdgcn_ballot_w32(m);
    if (m) loc1[t] = run + __popc(bal & below_mask);
    run += __popc(bal);
  }
  if (lane == 0) { splits_out[e] = (float)run; splits1_ws[e] = run; }
  const int s1 = run;
  int run2 = 0;
  for (int base = 0; base < T_TOK; base += 32) {
    int t = base + lane;
    bool m = (idx2[t] == e);
    unsigned bal = __builtin_amdgcn_ballot_w32(m);
    if (m) loc2[t] = s1 + run2 + __popc(bal & below_mask);
    run2 += __popc(bal);
  }
  if (lane == 0) splits_out[NEXP + e] = (float)run2;
}

// ---------------------------------------------------------------------------
// K3: zero-fill combine+dispatch (the 1 GB roofline term). Non-temporal B128
// stores; scalar head/tail since combine starts at out+1 (4B misaligned).
// ---------------------------------------------------------------------------
__global__ void zero_kernel(float* __restrict__ base, long lo, long hi) {
  long alo = (lo + 3) & ~3L;
  long ahi = hi & ~3L;
  long nv  = (ahi - alo) >> 2;
  long i      = blockIdx.x * (long)blockDim.x + threadIdx.x;
  long stride = (long)gridDim.x * blockDim.x;
  v4f z = {0.f, 0.f, 0.f, 0.f};
  for (long v = i; v < nv; v += stride)
    __builtin_nontemporal_store(z, (v4f*)(base + alo + v * 4));
  if (i == 0) {
    for (long p = lo;  p < alo; ++p) base[p] = 0.f;
    for (long p = ahi; p < hi;  ++p) base[p] = 0.f;
  }
}

// ---------------------------------------------------------------------------
// K4: scatter the <=2 nonzeros per token with post-drop gate normalization.
// ---------------------------------------------------------------------------
__global__ void scatter_kernel(
    const int* __restrict__ idx1, const int* __restrict__ idx2,
    const int* __restrict__ loc1, const int* __restrict__ loc2,
    const float* __restrict__ g1raw, const float* __restrict__ g2raw,
    float* __restrict__ combine, float* __restrict__ dispatch) {
  int t = blockIdx.x * blockDim.x + threadIdx.x;
  if (t >= T_TOK) return;
  int e1 = idx1[t], e2 = idx2[t], l1 = loc1[t], l2 = loc2[t];
  bool keep1 = l1 < CAPC, keep2 = l2 < CAPC;
  float g1 = keep1 ? g1raw[t] : 0.f;
  float g2 = keep2 ? g2raw[t] : 0.f;
  float den = fmaxf(g1 + g2, FLT_EPSILON);   // jnp.finfo(f32).eps
  g1 /= den; g2 /= den;
  size_t b = (size_t)t * NEXP * CAPC;
  if (keep1) {
    size_t o = b + (size_t)e1 * CAPC + l1;
    combine[o] = g1; dispatch[o] = (g1 > 0.f) ? 1.f : 0.f;
  }
  if (keep2) {
    size_t o = b + (size_t)e2 * CAPC + l2;
    combine[o] = g2; dispatch[o] = (g2 > 0.f) ? 1.f : 0.f;
  }
}

// ---------------------------------------------------------------------------
// K5: l_aux = E * sum_e (me_sum[e]/T) * (splits1[e]/T)
// ---------------------------------------------------------------------------
__global__ __launch_bounds__(64) void laux_kernel(const float* __restrict__ me_sum,
                                                  const int* __restrict__ splits1_ws,
                                                  float* __restrict__ out0) {
  __shared__ float red[NEXP];
  int e = threadIdx.x;
  float me = me_sum[e] * (1.f / (float)T_TOK);
  float ce = (float)splits1_ws[e] * (1.f / (float)T_TOK);
  red[e] = me * ce;
  __syncthreads();
  for (int off = 32; off > 0; off >>= 1) {
    if (e < off) red[e] += red[e + off];
    __syncthreads();
  }
  if (e == 0) out0[0] = red[0] * (float)NEXP;
}

// ---------------------------------------------------------------------------
extern "C" void kernel_launch(void* const* d_in, const int* in_sizes, int n_in,
                              void* d_out, int out_size, void* d_ws, size_t ws_size,
                              hipStream_t stream) {
  const float* x     = (const float*)d_in[0];   // [T, D]
  const float* w_red = (const float*)d_in[1];   // [4, D]
  const float* cent  = (const float*)d_in[2];   // [E, 4]
  float* out = (float*)d_out;
  char*  ws  = (char*)d_ws;

  _Float16* w2h  = (_Float16*)(ws + 0);        // 64*2048*2  = 262144 B
  int*   idx1    = (int*)  (ws + 262144);      // T ints
  int*   idx2    = (int*)  (ws + 294912);
  int*   loc1    = (int*)  (ws + 327680);
  int*   loc2    = (int*)  (ws + 360448);
  float* g1r     = (float*)(ws + 393216);
  float* g2r     = (float*)(ws + 425984);
  float* me      = (float*)(ws + 458752);      // [64]
  int*   s1ws    = (int*)  (ws + 459008);      // [64]

  const long TEC = (long)T_TOK * NEXP * CAPC;  // 134217728
  float* combine    = out + 1;
  float* dispatch   = out + 1 + TEC;
  float* splits_out = out + 1 + 2 * TEC;

  prep_kernel     <<<512, 256, 0, stream>>>(w_red, cent, w2h, me);
  gate_kernel     <<<128, 128, 0, stream>>>(x, w2h, idx1, idx2, g1r, g2r, me);
  positions_kernel<<< 64,  32, 0, stream>>>(idx1, idx2, loc1, loc2, splits_out, s1ws);
  zero_kernel     <<<4096,256, 0, stream>>>(out, 1L, 1L + 2 * TEC);
  scatter_kernel  <<< 32, 256, 0, stream>>>(idx1, idx2, loc1, loc2, g1r, g2r, combine, dispatch);
  laux_kernel     <<<  1,  64, 0, stream>>>(me, s1ws, out);
}
